// GIN_7997229105403
// MI455X (gfx1250) — matrix operand order repaired and checked by reading, compiled, not run on hardware
//
#include <hip/hip_runtime.h>
#include <hip/hip_bf16.h>

#define N_NODES 100000
#define N_EDGES 1600000
#define N_FEAT  128
#define N_HID   64
#define N_CLASS 10
#define N_GRAPHS 1000

#define PERSIST_BLOCKS 256   // persistent-wave grid for the WMMA kernels

typedef __attribute__((ext_vector_type(16))) __bf16 v16bf;
typedef __attribute__((ext_vector_type(8)))  __bf16 v8bf;
typedef __attribute__((ext_vector_type(8)))  float  v8f;

#define WMMA_BF16(a, b, c) \
  __builtin_amdgcn_wmma_f32_16x16x32_bf16(false, (a), false, (b), (short)0, (c), false, false)

// ---------------------------------------------------------------------------
// A-fragment (16x32 bf16) from row-major fp32, CDNA5 16-bit A layout:
//   lanes 0-15 : M=lane,    K = k0+{0..7} and k0+16+{0..7}
//   lanes16-31 : M=lane-16, K = k0+8+{0..7} and k0+24+{0..7}
// caller passes p = base + row*ld + k0 + hi*8  (two 32B-contiguous runs)
__device__ __forceinline__ v16bf load_a_f32(const float* p) {
  v16bf a;
#pragma unroll
  for (int i = 0; i < 8; ++i) a[i] = (__bf16)p[i];
#pragma unroll
  for (int i = 0; i < 8; ++i) a[8 + i] = (__bf16)p[16 + i];
  return a;
}

// A-fragment from bf16 LDS scratch: two contiguous 16B runs -> 2x ds_load_b128
__device__ __forceinline__ v16bf load_a_bf16(const __bf16* q) {
  union { v16bf v; v8bf h[2]; } u;
  u.h[0] = *(const v8bf*)(q);
  u.h[1] = *(const v8bf*)(q + 16);
  return u.v;
}

// B-fragment from N-major (transposed) bf16 weights in LDS:
//   lane holds col = lane%16, K = k0 + (lane/16)*8 + {0..7} and k0 + 16 + (lane/16)*8 + {0..7}
__device__ __forceinline__ v16bf load_bT(const __bf16* wT, int ldk, int k0, int col, int hi) {
  union { v16bf v; v8bf h[2]; } u;
  const __bf16* q = wT + col * ldk + k0 + hi * 8;
  u.h[0] = *(const v8bf*)(q);
  u.h[1] = *(const v8bf*)(q + 16);
  return u.v;
}

// ---------------------------------------------------------------------------
// h = x @ pre_w + pre_b   [100k x 128] @ [128 x 64]
// 8 waves/block, each wave owns a 32-row x 64-col tile (2 A frags share each B)
__global__ void __launch_bounds__(256, 1)
k_pre(const float* __restrict__ x, const float* __restrict__ w,
      const float* __restrict__ bias, float* __restrict__ h,
      int nRows, int nTiles) {
  __shared__ __bf16 wT[N_HID * N_FEAT];   // [n][k], 16 KB
  __shared__ float  bl[N_HID];
  for (int i = threadIdx.x; i < N_FEAT * N_HID; i += 256) {
    const int k = i >> 6, n = i & 63;     // w is [k][n] row-major
    wT[n * N_FEAT + k] = (__bf16)w[i];
  }
  if (threadIdx.x < N_HID) bl[threadIdx.x] = bias[threadIdx.x];
  __syncthreads();

  const int wave = threadIdx.x >> 5, lane = threadIdx.x & 31;
  const int hi = lane >> 4, l = lane & 15;

  const int stride = gridDim.x * 8;
  for (int t = blockIdx.x * 8 + wave; t < nTiles; t += stride) {
    const int r0 = t * 32;
    const int rowA0 = min(r0 + l,      nRows - 1);
    const int rowA1 = min(r0 + 16 + l, nRows - 1);
    v8f c[2][4] = {};
#pragma unroll
    for (int ks = 0; ks < 4; ++ks) {
      const v16bf a0 = load_a_f32(x + (long)rowA0 * N_FEAT + ks * 32 + hi * 8);
      const v16bf a1 = load_a_f32(x + (long)rowA1 * N_FEAT + ks * 32 + hi * 8);
#pragma unroll
      for (int j = 0; j < 4; ++j) {
        const v16bf b = load_bT(wT, N_FEAT, ks * 32, j * 16 + l, hi);
        c[0][j] = WMMA_BF16(a0, b, c[0][j]);
        c[1][j] = WMMA_BF16(a1, b, c[1][j]);
      }
    }
#pragma unroll
    for (int half = 0; half < 2; ++half)
#pragma unroll
      for (int v = 0; v < 8; ++v) {
        const int row = r0 + half * 16 + v + hi * 8;
        if (row < nRows) {
          float* o = h + (long)row * N_HID;
#pragma unroll
          for (int j = 0; j < 4; ++j) o[j * 16 + l] = c[half][j][v] + bl[j * 16 + l];
        }
      }
  }
}

// ---------------------------------------------------------------------------
// agg[dst] += h[src], 16 threads per edge, float4 per thread; h is L2-resident
__global__ void k_edge_agg(const float* __restrict__ h, const long long* __restrict__ src,
                           const long long* __restrict__ dst, float* __restrict__ agg,
                           int nEdges) {
  const int gid = blockIdx.x * blockDim.x + threadIdx.x;
  const int e = gid >> 4;
  if (e >= nEdges) return;
  const int f = (gid & 15) * 4;
  if (f == 0 && e + 64 < nEdges)                       // warm the gather stream
    __builtin_prefetch(h + src[e + 64] * N_HID, 0, 0); // -> global_prefetch_b8
  const long long s = src[e], d = dst[e];
  const float4 v = *(const float4*)(h + s * N_HID + f);
  float* o = agg + d * N_HID + f;
  atomicAdd(o + 0, v.x);
  atomicAdd(o + 1, v.y);
  atomicAdd(o + 2, v.z);
  atomicAdd(o + 3, v.w);
}

// ---------------------------------------------------------------------------
// h = relu( relu((h+agg) @ w1 + b1) @ w2 + b2 )  in place; 32-row wave tiles
__global__ void __launch_bounds__(256, 1)
k_mlp(const float* __restrict__ hin, const float* __restrict__ agg,
      const float* __restrict__ w1, const float* __restrict__ b1,
      const float* __restrict__ w2, const float* __restrict__ b2,
      float* __restrict__ hout, int nRows, int nTiles) {
  __shared__ __bf16 w1T[N_HID * N_HID];   // [n][k], 8 KB each
  __shared__ __bf16 w2T[N_HID * N_HID];
  __shared__ float  b1l[N_HID], b2l[N_HID];
  __shared__ __bf16 mid[8][32 * N_HID];   // per-wave C->A bounce, 32 KB total
  for (int i = threadIdx.x; i < N_HID * N_HID; i += 256) {
    const int k = i >> 6, n = i & 63;
    w1T[n * N_HID + k] = (__bf16)w1[i];
    w2T[n * N_HID + k] = (__bf16)w2[i];
  }
  if (threadIdx.x < N_HID) { b1l[threadIdx.x] = b1[threadIdx.x]; b2l[threadIdx.x] = b2[threadIdx.x]; }
  __syncthreads();

  const int wave = threadIdx.x >> 5, lane = threadIdx.x & 31;
  const int hi = lane >> 4, l = lane & 15;

  __bf16* ms = mid[wave];
  const int stride = gridDim.x * 8;
  for (int t = blockIdx.x * 8 + wave; t < nTiles; t += stride) {
    const int r0 = t * 32;
    const int rowA0 = min(r0 + l,      nRows - 1);
    const int rowA1 = min(r0 + 16 + l, nRows - 1);

    // ---- GEMM1: (h+agg) @ w1
    v8f c[2][4] = {};
#pragma unroll
    for (int ks = 0; ks < 2; ++ks) {
      v16bf a0, a1;
      {
        const float* p1 = hin + (long)rowA0 * N_HID + ks * 32 + hi * 8;
        const float* p2 = agg + (long)rowA0 * N_HID + ks * 32 + hi * 8;
#pragma unroll
        for (int i = 0; i < 8; ++i) a0[i]     = (__bf16)(p1[i]      + p2[i]);
#pragma unroll
        for (int i = 0; i < 8; ++i) a0[8 + i] = (__bf16)(p1[16 + i] + p2[16 + i]);
      }
      {
        const float* p1 = hin + (long)rowA1 * N_HID + ks * 32 + hi * 8;
        const float* p2 = agg + (long)rowA1 * N_HID + ks * 32 + hi * 8;
#pragma unroll
        for (int i = 0; i < 8; ++i) a1[i]     = (__bf16)(p1[i]      + p2[i]);
#pragma unroll
        for (int i = 0; i < 8; ++i) a1[8 + i] = (__bf16)(p1[16 + i] + p2[16 + i]);
      }
#pragma unroll
      for (int j = 0; j < 4; ++j) {
        const v16bf b = load_bT(w1T, N_HID, ks * 32, j * 16 + l, hi);
        c[0][j] = WMMA_BF16(a0, b, c[0][j]);
        c[1][j] = WMMA_BF16(a1, b, c[1][j]);
      }
    }

    // ---- bias + relu, bounce C-layout -> A-layout through LDS (bf16)
#pragma unroll
    for (int half = 0; half < 2; ++half)
#pragma unroll
      for (int j = 0; j < 4; ++j)
#pragma unroll
        for (int v = 0; v < 8; ++v) {
          const float z = c[half][j][v] + b1l[j * 16 + l];
          ms[(half * 16 + v + hi * 8) * N_HID + j * 16 + l] = (__bf16)fmaxf(z, 0.f);
        }
    // cross-lane LDS RAW within the wave: order ds write -> ds read
    asm volatile("s_wait_dscnt 0" ::: "memory");

    // ---- GEMM2: relu(.) @ w2
    v8f d[2][4] = {};
#pragma unroll
    for (int ks = 0; ks < 2; ++ks) {
      const v16bf a0 = load_a_bf16(ms + (l)      * N_HID + ks * 32 + hi * 8);
      const v16bf a1 = load_a_bf16(ms + (16 + l) * N_HID + ks * 32 + hi * 8);
#pragma unroll
      for (int j = 0; j < 4; ++j) {
        const v16bf b = load_bT(w2T, N_HID, ks * 32, j * 16 + l, hi);
        d[0][j] = WMMA_BF16(a0, b, d[0][j]);
        d[1][j] = WMMA_BF16(a1, b, d[1][j]);
      }
    }
    // WAR: keep next iteration's bounce writes behind this iteration's reads
    asm volatile("s_wait_dscnt 0" ::: "memory");

    // ---- bias + relu + store
#pragma unroll
    for (int half = 0; half < 2; ++half)
#pragma unroll
      for (int v = 0; v < 8; ++v) {
        const int row = r0 + half * 16 + v + hi * 8;
        if (row < nRows) {
          float* o = hout + (long)row * N_HID;
#pragma unroll
          for (int j = 0; j < 4; ++j)
            o[j * 16 + l] = fmaxf(d[half][j][v] + b2l[j * 16 + l], 0.f);
        }
      }
  }
}

// ---------------------------------------------------------------------------
// g[batch[n]] += h[n], 16 threads per node
__global__ void k_pool(const float* __restrict__ h, const long long* __restrict__ batch,
                       float* __restrict__ g, int nRows) {
  const int gid = blockIdx.x * blockDim.x + threadIdx.x;
  const int n = gid >> 4;
  if (n >= nRows) return;
  const int f = (gid & 15) * 4;
  const long long b = batch[n];
  const float4 v = *(const float4*)(h + (long)n * N_HID + f);
  float* o = g + b * N_HID + f;
  atomicAdd(o + 0, v.x);
  atomicAdd(o + 1, v.y);
  atomicAdd(o + 2, v.z);
  atomicAdd(o + 3, v.w);
}

// ---------------------------------------------------------------------------
// out = log_softmax(relu(g @ post_w + post_b) @ ro_w + ro_b); one block/graph
__global__ void k_head(const float* __restrict__ g, const float* __restrict__ pw,
                       const float* __restrict__ pb, const float* __restrict__ rw,
                       const float* __restrict__ rb, float* __restrict__ out) {
  __shared__ float hbuf[N_HID];
  __shared__ float logits[N_CLASS];
  __shared__ float lse;
  const int gi = blockIdx.x;
  const int t  = threadIdx.x;

  float acc = pb[t];
  const float* gr = g + gi * N_HID;
#pragma unroll 8
  for (int k = 0; k < N_HID; ++k) acc += gr[k] * pw[k * N_HID + t];
  hbuf[t] = fmaxf(acc, 0.f);
  __syncthreads();

  if (t < N_CLASS) {
    float a = rb[t];
#pragma unroll 8
    for (int k = 0; k < N_HID; ++k) a += hbuf[k] * rw[k * N_CLASS + t];
    logits[t] = a;
  }
  __syncthreads();

  if (t == 0) {
    float m = logits[0];
    for (int c = 1; c < N_CLASS; ++c) m = fmaxf(m, logits[c]);
    float s = 0.f;
    for (int c = 0; c < N_CLASS; ++c) s += __expf(logits[c] - m);
    lse = m + __logf(s);
  }
  __syncthreads();

  if (t < N_CLASS) out[gi * N_CLASS + t] = logits[t] - lse;
}

// ---------------------------------------------------------------------------
extern "C" void kernel_launch(void* const* d_in, const int* in_sizes, int n_in,
                              void* d_out, int out_size, void* d_ws, size_t ws_size,
                              hipStream_t stream) {
  const float*     x       = (const float*)d_in[0];
  const long long* ei      = (const long long*)d_in[1];   // [2, E] int64
  const long long* batch   = (const long long*)d_in[2];
  const float*     pre_w   = (const float*)d_in[3];
  const float*     pre_b   = (const float*)d_in[4];
  const float*     conv_w1 = (const float*)d_in[5];       // [3,64,64]
  const float*     conv_b1 = (const float*)d_in[6];
  const float*     conv_w2 = (const float*)d_in[7];
  const float*     conv_b2 = (const float*)d_in[8];
  const float*     post_w  = (const float*)d_in[9];
  const float*     post_b  = (const float*)d_in[10];
  const float*     ro_w    = (const float*)d_in[11];
  const float*     ro_b    = (const float*)d_in[12];
  float* out = (float*)d_out;

  float* h   = (float*)d_ws;                   // [N_NODES * 64]
  float* agg = h + (size_t)N_NODES * N_HID;    // [N_NODES * 64]
  float* g   = agg + (size_t)N_NODES * N_HID;  // [N_GRAPHS * 64]

  const long long* src = ei;
  const long long* dst = ei + N_EDGES;

  const int nTiles     = (N_NODES + 31) / 32;                  // 3125 (32-row tiles)
  const int edgeBlocks = ((N_EDGES * 16) + 255) / 256;
  const int poolBlocks = ((N_NODES * 16) + 255) / 256;

  k_pre<<<PERSIST_BLOCKS, 256, 0, stream>>>(x, pre_w, pre_b, h, N_NODES, nTiles);

  for (int l = 0; l < 3; ++l) {
    hipMemsetAsync(agg, 0, (size_t)N_NODES * N_HID * sizeof(float), stream);
    k_edge_agg<<<edgeBlocks, 256, 0, stream>>>(h, src, dst, agg, N_EDGES);
    k_mlp<<<PERSIST_BLOCKS, 256, 0, stream>>>(h, agg,
                                              conv_w1 + (size_t)l * N_HID * N_HID,
                                              conv_b1 + (size_t)l * N_HID,
                                              conv_w2 + (size_t)l * N_HID * N_HID,
                                              conv_b2 + (size_t)l * N_HID,
                                              h, N_NODES, nTiles);
  }

  hipMemsetAsync(g, 0, (size_t)N_GRAPHS * N_HID * sizeof(float), stream);
  k_pool<<<poolBlocks, 256, 0, stream>>>(h, batch, g, N_NODES);
  k_head<<<N_GRAPHS, N_HID, 0, stream>>>(g, post_w, post_b, ro_w, ro_b, out);
}
